// VectorQuantizer5_34703335751962
// MI455X (gfx1250) — matrix operand order repaired and checked by reading, not compiled, not run on hardware
//
#include <hip/hip_runtime.h>

// ---------------------------------------------------------------------------
// VQ-VAE vector quantizer forward for gfx1250 (MI455X).
// All GEMMs (2 convs as implicit GEMM + codebook distances) run on
// v_wmma_f32_16x16x32_bf16 (wave32). Conv input patches are staged into LDS
// (via async-to-LDS when available) and shared by all 8 waves of a workgroup.
//   B=32, C=256, H=W=32, N_E=1024, E_DIM=256, BETA=0.25
// ---------------------------------------------------------------------------

typedef __attribute__((ext_vector_type(16))) __bf16 v16bf;
typedef __attribute__((ext_vector_type(8)))  __bf16 v8bf;
typedef __attribute__((ext_vector_type(8)))  float  v8f;
typedef __attribute__((ext_vector_type(4)))  int    v4i;

#define NPIX   32768          // B*H*W
#define EDIM   256
#define NCODE  1024
#define LOSS_SCALE (1.25f / 8388608.0f)   // (1+beta) / (NPIX*EDIM)

// CDNA5 async global->LDS path (ASYNCcnt-tracked), with safe fallback.
// NOTE: builtin params are NON-const __device__/__shared__ pointers.
#if __has_builtin(__builtin_amdgcn_global_load_async_to_lds_b128) && \
    __has_builtin(__builtin_amdgcn_s_wait_asynccnt)
#define USE_ASYNC_LDS 1
typedef __attribute__((address_space(1))) v4i* gp_v4i;
typedef __attribute__((address_space(3))) v4i* lp_v4i;
#else
#define USE_ASYNC_LDS 0
#endif

// ---- workspace layout (bytes) ----
#define OFF_ZB   ((size_t)0)          // z NHWC bf16        16777216
#define OFF_WT1  ((size_t)16777216)   // emb_w  [9][O][I]   1179648
#define OFF_WT2  ((size_t)17956864)   // unemb_w [9][O][I]  1179648
#define OFF_CBBF ((size_t)19136512)   // codebook bf16       524288
#define OFF_CBN  ((size_t)19660800)   // codebook norms        4096
#define OFF_ZEF  ((size_t)19664896)   // ze f32 NHWC       33554432
#define OFF_ZEB  ((size_t)53219328)   // ze bf16 NHWC      16777216
#define OFF_ZQB  ((size_t)69996544)   // z_q bf16 NHWC     16777216

// ---------------------------------------------------------------------------
// Fragment helpers (contiguous k order used by this file for A and B, bf16 16x16x32)
// A (16x32): lane m = lane&15; half g = lane>>4 supplies
//            K in {16g..16g+15} (elems 0..15)
// B (32x16): lane col n = lane&15; half g supplies K in {16g..16g+15}
// ---------------------------------------------------------------------------
__device__ __forceinline__ v16bf load_a_pair(const __bf16* base, int c0, int g) {
  v16bf a;
  v8bf lo = *(const v8bf*)(base + c0 + 16 * g);
  v8bf hi = *(const v8bf*)(base + c0 + 16 * g + 8);
#pragma unroll
  for (int i = 0; i < 8; ++i) { a[i] = lo[i]; a[i + 8] = hi[i]; }
  return a;
}

__device__ __forceinline__ v16bf load_b_frag(const __bf16* p) {
  v16bf b;
  v8bf lo = *(const v8bf*)(p);
  v8bf hi = *(const v8bf*)(p + 8);
#pragma unroll
  for (int i = 0; i < 8; ++i) { b[i] = lo[i]; b[i + 8] = hi[i]; }
  return b;
}

// ---------------------------------------------------------------------------
// Prep kernels
// ---------------------------------------------------------------------------
__global__ void k_prep_z(const float* __restrict__ z, __bf16* __restrict__ zb) {
  int i = blockIdx.x * 256 + threadIdx.x;           // NHWC flat index
  int c = i & 255;
  int x = (i >> 8) & 31;
  int y = (i >> 13) & 31;
  int b = i >> 18;
  zb[i] = (__bf16)z[(((size_t)b * 256 + c) * 32 + y) * 32 + x];
}

__global__ void k_prep_w(const float* __restrict__ w, __bf16* __restrict__ wt) {
  // w: [O=256][I=256][3][3]  ->  wt: [tap=9][o][ci]
  int i = blockIdx.x * 256 + threadIdx.x;           // over 9*256*256
  int ci  = i & 255;
  int o   = (i >> 8) & 255;
  int tap = i >> 16;
  wt[i] = (__bf16)w[((size_t)o * 256 + ci) * 9 + tap];
}

__global__ void k_prep_cb(const float* __restrict__ cb, __bf16* __restrict__ cbf,
                          float* __restrict__ cbn, float* __restrict__ loss_slot) {
  __shared__ float red[256];
  int n = blockIdx.x, d = threadIdx.x;
  float v = cb[(size_t)n * 256 + d];
  cbf[(size_t)n * 256 + d] = (__bf16)v;
  red[d] = v * v;
  __syncthreads();
#pragma unroll
  for (int s = 128; s > 0; s >>= 1) {
    if (d < s) red[d] += red[d + s];
    __syncthreads();
  }
  if (d == 0) cbn[n] = red[0];
  if (n == 0 && d == 0) *loss_slot = 0.0f;          // re-zero every launch
}

// ---------------------------------------------------------------------------
// 3x3 SAME conv as implicit GEMM.
// One workgroup (8 waves) = one image row: 32 pixels x 256 out-channels.
// Input patch 3x34x256 bf16 (51KB) staged once in LDS (padding pre-zeroed),
// shared by all waves. Each wave: 2 pixel-halves x 2 channel-tiles
// (4 accumulators) -> each A/B fragment feeds 2 WMMAs.
// ---------------------------------------------------------------------------
__global__ __launch_bounds__(256) void k_conv_wmma(
    const __bf16* __restrict__ x,       // NHWC bf16 input
    const __bf16* __restrict__ wt,      // [9][o][ci] bf16
    const float*  __restrict__ bias,    // [256]
    float*  __restrict__ out_nhwc_f32,  // may be null
    __bf16* __restrict__ out_nhwc_bf,   // may be null
    float*  __restrict__ out_nchw_f32)  // may be null
{
  __shared__ __bf16 patch[3 * 34 * 256];            // 52224 B

  const int tid = threadIdx.x;
  const int mRow = blockIdx.x;                      // b*32 + h
  const int h = mRow & 31;
  const int b = mRow >> 5;

  // ---- stage input patch (rows h-1..h+1, x -1..32, all 256 ch) into LDS ----
  for (int i = tid; i < 3264; i += 256) {           // 3264 chunks of 8 bf16
    const int slot = i >> 5;                        // 0..101 = ry*34+rx
    const int ch   = (i & 31) * 8;
    const int ry = slot / 34, rx = slot % 34;
    const int y = h + ry - 1, xx = rx - 1;
    const bool valid = (y >= 0) & (y < 32) & (xx >= 0) & (xx < 32);
    __bf16* dst = patch + (size_t)i * 8;
    if (valid) {
      const __bf16* src = x + ((size_t)((b * 32 + y) * 32 + xx)) * EDIM + ch;
#if USE_ASYNC_LDS
      __builtin_amdgcn_global_load_async_to_lds_b128(
          (gp_v4i)(unsigned long long)src, (lp_v4i)(unsigned long long)dst, 0, 0);
#else
      *(v8bf*)dst = *(const v8bf*)src;
#endif
    } else {
      v8bf zz = {};
      *(v8bf*)dst = zz;
    }
  }
#if USE_ASYNC_LDS
  __builtin_amdgcn_s_wait_asynccnt(0);
#endif
  __syncthreads();

  // ---- compute ----
  const int wave = tid >> 5;
  const int lane = tid & 31;
  const int g = lane >> 4;
  const int n = lane & 15;
  const int oB0 = wave * 32 + n;                    // this lane's B columns
  const int oB1 = oB0 + 16;

  v8f acc00 = {}, acc01 = {}, acc10 = {}, acc11 = {};
  for (int ky = 0; ky < 3; ++ky) {
    for (int kx = 0; kx < 3; ++kx) {
      const __bf16* arow0 = patch + (size_t)(ky * 34 + n + kx) * EDIM;
      const __bf16* arow1 = arow0 + 16 * EDIM;
      const __bf16* bp0 = wt + ((size_t)((ky * 3 + kx) * 256) + oB0) * EDIM
                             + 16 * g;
      const __bf16* bp1 = bp0 + 16 * EDIM;
      __builtin_prefetch(bp0 + 9 * EDIM, 0, 1);     // warm weight row 9 out-channels ahead (same tap)
#pragma unroll
      for (int cs = 0; cs < 8; ++cs) {
        const int c0 = cs * 32;
        v16bf a0 = load_a_pair(arow0, c0, g);
        v16bf a1 = load_a_pair(arow1, c0, g);
        v16bf b0 = load_b_frag(bp0 + c0);
        v16bf b1 = load_b_frag(bp1 + c0);
        acc00 = __builtin_amdgcn_wmma_f32_16x16x32_bf16(
            false, a0, false, b0, (short)0, acc00, false, false);
        acc01 = __builtin_amdgcn_wmma_f32_16x16x32_bf16(
            false, a0, false, b1, (short)0, acc01, false, false);
        acc10 = __builtin_amdgcn_wmma_f32_16x16x32_bf16(
            false, a1, false, b0, (short)0, acc10, false, false);
        acc11 = __builtin_amdgcn_wmma_f32_16x16x32_bf16(
            false, a1, false, b1, (short)0, acc11, false, false);
      }
    }
  }

  // ---- store 4 tiles ----
  const float bv0 = bias[oB0];
  const float bv1 = bias[oB1];
  const int pBase = mRow * 32;
#pragma unroll
  for (int t = 0; t < 4; ++t) {
    const v8f acc = (t == 0) ? acc00 : (t == 1) ? acc01 : (t == 2) ? acc10 : acc11;
    const int pB = pBase + (t >> 1) * 16;
    const int o  = (t & 1) ? oB1 : oB0;
    const float bv = (t & 1) ? bv1 : bv0;
#pragma unroll
    for (int v = 0; v < 8; ++v) {
      const int p = pB + v + 8 * g;                 // D row = v + 8*(lane>>4)
      const float val = acc[v] + bv;
      if (out_nhwc_f32) out_nhwc_f32[(size_t)p * EDIM + o] = val;
      if (out_nhwc_bf)  out_nhwc_bf[(size_t)p * EDIM + o] = (__bf16)val;
      if (out_nchw_f32) {
        const int xw = p & 31, yh = (p >> 5) & 31, bb = p >> 10;
        out_nchw_f32[(((size_t)bb * 256 + o) * 32 + yh) * 32 + xw] = val;
      }
    }
  }
}

// ---------------------------------------------------------------------------
// Codebook distance GEMM + argmin + gather + loss. One wave per 16 pixels.
// score = ||e_n||^2 - 2*z.e   (||z||^2 constant per row -> argmin-invariant)
// ---------------------------------------------------------------------------
__global__ __launch_bounds__(128) void k_vq_argmin(
    const __bf16* __restrict__ ze_bf, const float* __restrict__ ze_f32,
    const __bf16* __restrict__ cb_bf, const float* __restrict__ cbn,
    const float* __restrict__ cb_f32,
    int* __restrict__ idx_out, float* __restrict__ loss_out,
    __bf16* __restrict__ zq_bf)
{
  const int wave = threadIdx.x >> 5;
  const int lane = threadIdx.x & 31;
  const int mTile = blockIdx.x * 4 + wave;
  const int pBase = mTile * 16;
  const int g = lane >> 4;
  const int n = lane & 15;

  // Cache the full K=256 A strip (8 fragments) in registers.
  v16bf afrag[8];
  const __bf16* ap = ze_bf + (size_t)(pBase + n) * EDIM;
#pragma unroll
  for (int cs = 0; cs < 8; ++cs)
    afrag[cs] = load_a_pair(ap, cs * 32, g);

  float minVal[8];
  int   minIdx[8];
#pragma unroll
  for (int v = 0; v < 8; ++v) { minVal[v] = 3.4e38f; minIdx[v] = 0; }

  for (int nt = 0; nt < 64; ++nt) {
    const int nb = nt * 16 + n;                     // this lane's column
    const __bf16* bp = cb_bf + (size_t)nb * EDIM + 16 * g;
    if (nt < 63) __builtin_prefetch(bp + 16 * EDIM, 0, 1);
    v8f acc = {};
#pragma unroll
    for (int cs = 0; cs < 8; ++cs) {
      v16bf bf = load_b_frag(bp + cs * 32);
      acc = __builtin_amdgcn_wmma_f32_16x16x32_bf16(
          false, afrag[cs], false, bf, (short)0, acc, false, false);
    }
    const float cn = cbn[nb];
#pragma unroll
    for (int v = 0; v < 8; ++v) {
      const float score = cn - 2.0f * acc[v];
      if (score < minVal[v]) { minVal[v] = score; minIdx[v] = nb; }
    }
  }

  // Butterfly min-reduce across the 16 lanes of each half (columns).
  // Tie-break on lower index to match jnp.argmin (first occurrence).
#pragma unroll
  for (int v = 0; v < 8; ++v) {
#pragma unroll
    for (int m = 8; m >= 1; m >>= 1) {
      const float ov = __shfl_xor(minVal[v], m, 32);
      const int   oi = __shfl_xor(minIdx[v], m, 32);
      if (ov < minVal[v] || (ov == minVal[v] && oi < minIdx[v])) {
        minVal[v] = ov; minIdx[v] = oi;
      }
    }
  }
  if (n == 0) {                       // lanes 0 and 16 own rows 8g..8g+7
#pragma unroll
    for (int v = 0; v < 8; ++v) idx_out[pBase + 8 * g + v] = minIdx[v];
  }

  // Gather f32 codebook rows, accumulate loss, emit z_q bf16.
  float lossAcc = 0.0f;
#pragma unroll
  for (int r = 0; r < 16; ++r) {
    const int src = (r < 8) ? 0 : 16;
    const int id = __shfl(minIdx[r & 7], src, 32);
    const float* crow = cb_f32 + (size_t)id * EDIM;
    const float* zrow = ze_f32 + (size_t)(pBase + r) * EDIM;
    __bf16* qrow = zq_bf + (size_t)(pBase + r) * EDIM;
#pragma unroll
    for (int d = 0; d < 8; ++d) {
      const int dd = lane * 8 + d;
      const float cv = crow[dd];
      const float df = cv - zrow[dd];
      lossAcc += df * df;
      qrow[dd] = (__bf16)cv;
    }
  }
#pragma unroll
  for (int m = 16; m >= 1; m >>= 1) lossAcc += __shfl_xor(lossAcc, m, 32);
  if (lane == 0) atomicAdd(loss_out, lossAcc * LOSS_SCALE);
}

// ---------------------------------------------------------------------------
// Launcher
// ---------------------------------------------------------------------------
extern "C" void kernel_launch(void* const* d_in, const int* in_sizes, int n_in,
                              void* d_out, int out_size, void* d_ws, size_t ws_size,
                              hipStream_t stream) {
  const float* z        = (const float*)d_in[0];
  const float* emb_w    = (const float*)d_in[1];
  const float* emb_b    = (const float*)d_in[2];
  const float* codebook = (const float*)d_in[3];
  const float* unemb_w  = (const float*)d_in[4];
  const float* unemb_b  = (const float*)d_in[5];

  char* ws = (char*)d_ws;
  __bf16* zb    = (__bf16*)(ws + OFF_ZB);
  __bf16* wt1   = (__bf16*)(ws + OFF_WT1);
  __bf16* wt2   = (__bf16*)(ws + OFF_WT2);
  __bf16* cbbf  = (__bf16*)(ws + OFF_CBBF);
  float*  cbn   = (float*)(ws + OFF_CBN);
  float*  zef   = (float*)(ws + OFF_ZEF);
  __bf16* zeb   = (__bf16*)(ws + OFF_ZEB);
  __bf16* zqb   = (__bf16*)(ws + OFF_ZQB);

  float* out  = (float*)d_out;            // [32,256,32,32] NCHW
  float* loss = out + 8388608;            // scalar
  int*   idx  = (int*)(out + 8388609);    // [32768] int32

  // Stage 0: precision/layout prep (+ zero loss slot).
  k_prep_z <<<NPIX * EDIM / 256, 256, 0, stream>>>(z, zb);
  k_prep_w <<<9 * 256, 256, 0, stream>>>(emb_w, wt1);
  k_prep_w <<<9 * 256, 256, 0, stream>>>(unemb_w, wt2);
  k_prep_cb<<<NCODE, 256, 0, stream>>>(codebook, cbbf, cbn, loss);

  // Stage 1: embed conv (implicit GEMM, WMMA bf16, LDS-staged input rows).
  k_conv_wmma<<<NPIX / 32, 256, 0, stream>>>(zb, wt1, emb_b, zef, zeb, nullptr);

  // Stage 2: codebook distances + argmin + gather + loss.
  k_vq_argmin<<<(NPIX / 16) / 4, 128, 0, stream>>>(
      zeb, zef, cbbf, cbn, codebook, idx, loss, zqb);

  // Stage 3: unembed conv -> NCHW f32 output.
  k_conv_wmma<<<NPIX / 32, 256, 0, stream>>>(zqb, wt2, unemb_b,
                                             nullptr, nullptr, out);
}
